// StandardPointHead_27728308862989
// MI455X (gfx1250) — matrix operand order repaired and checked by reading, not compiled, not run on hardware
//
#include <hip/hip_runtime.h>
#include <hip/hip_bf16.h>

typedef __attribute__((ext_vector_type(16))) __bf16 v16bf;
typedef __attribute__((ext_vector_type(8)))  __bf16 v8bf;
typedef __attribute__((ext_vector_type(4)))  __bf16 v4bf;
typedef __attribute__((ext_vector_type(8)))  float  v8f;

// Problem constants (fixed by setup_inputs)
#define BATCH 2
#define CH    256
#define HH    128
#define WW    128
#define QQ    32768
#define KCONV 2304      // 9 taps * 256 cin
#define NCONV 512       // coef(256) ++ freq(256)

// ---------------------------------------------------------------------------
// Kernel 1: pack weights into WMMA-B friendly layouts (bf16, N-major rows)
//   Wn [512][2304]  : conv weights, k = tap*256 + cin
//   Wm [3][256][256]: MLP weights transposed, Wm[l][n][k] = w_l[k][n]
// ---------------------------------------------------------------------------
__global__ void ph_pack_weights(const float* __restrict__ coef_w,
                                const float* __restrict__ freq_w,
                                const float* __restrict__ w0,
                                const float* __restrict__ w1,
                                const float* __restrict__ w2,
                                __bf16* __restrict__ Wn,
                                __bf16* __restrict__ Wm) {
    int tid = blockIdx.x * blockDim.x + threadIdx.x;
    const int totWn = NCONV * KCONV;          // 1,179,648
    const int totWm = 3 * 256 * 256;          //   196,608
    if (tid < totWn) {
        int n = tid / KCONV, k = tid % KCONV;
        int tap = k / CH, cin = k % CH;
        int dy = tap / 3, dx = tap % 3;
        float v;
        if (n < 256) v = coef_w[(((n * CH) + cin) * 3 + dy) * 3 + dx];
        else         v = freq_w[((((n - 256) * CH) + cin) * 3 + dy) * 3 + dx];
        Wn[tid] = (__bf16)v;
    } else if (tid < totWn + totWm) {
        int t = tid - totWn;
        int layer = t >> 16;
        int r = t & 65535;
        int n = r >> 8, k = r & 255;
        const float* w = (layer == 0) ? w0 : (layer == 1) ? w1 : w2;
        Wm[t] = (__bf16)w[k * 256 + n];       // transpose to [n][k]
    }
}

// ---------------------------------------------------------------------------
// Kernel 2: feat [B][C][H][W] f32 -> feat_t [B][H][W][C] bf16
// ---------------------------------------------------------------------------
__global__ void ph_feat_t(const float* __restrict__ feat, __bf16* __restrict__ ft) {
    int i = blockIdx.x * 256 + threadIdx.x;   // 8,388,608 total
    int c = i & 255;
    int s = i >> 8;                            // b*16384 + y*128 + x
    int b = s >> 14;
    int yx = s & 16383;
    ft[i] = (__bf16)feat[((b * CH + c) << 14) + yx];
}

// ---------------------------------------------------------------------------
// Kernel 3: fused coef+freq 3x3 conv as implicit GEMM with bf16 WMMA.
//   block = (b, y, xtile): 16 spatial x 512 out-ch, 4 waves (128 N each),
//   each wave: 8 acc tiles, loop 9 taps x 8 K-chunks -> 576 WMMAs.
//   Output cf [B][H][W][512] bf16 (gather-friendly: 1KB per pixel).
// ---------------------------------------------------------------------------
__global__ void __launch_bounds__(128) ph_conv(const __bf16* __restrict__ ft,
                                               const __bf16* __restrict__ Wn,
                                               const float* __restrict__ coef_b,
                                               const float* __restrict__ freq_b,
                                               __bf16* __restrict__ cf) {
    const int wave = threadIdx.x >> 5;
    const int lane = threadIdx.x & 31;
    const int blk  = blockIdx.x;              // 0..2047
    const int xt = blk & 7;
    const int y  = (blk >> 3) & 127;
    const int b  = blk >> 10;
    const int x0 = xt << 4;
    const int m  = lane & 15;
    const int kb  = (lane < 16) ? 0 : 16;     // A-frag K base (contiguous K layout)
    const int kb2 = (lane < 16) ? 0 : 16;     // B-frag K base (dense 32x16 B layout)
    const int n0 = wave << 7;                 // wave's 128-wide N slab
    const int xA = x0 + m;                    // this lane's A row (image x)

    v8f acc[8];
#pragma unroll
    for (int i = 0; i < 8; ++i) acc[i] = (v8f){};

    for (int dy = 0; dy < 3; ++dy) {
        const int yy = y + dy - 1;
        const bool rowok = (yy >= 0) & (yy < HH);
        for (int dx = 0; dx < 3; ++dx) {
            const int xx = xA + dx - 1;
            const bool ok = rowok & (xx >= 0) & (xx < WW);
            const int abase = ((b << 14) + (yy << 7) + xx) << 8;   // *C
            const int tap = dy * 3 + dx;
            const __bf16* wt = Wn + tap * CH + kb2;
            for (int kc = 0; kc < 8; ++kc) {
                const int c0 = kc << 5;
                v16bf a = {};
                if (ok) {
                    union { v16bf v; v8bf h[2]; } u;
                    u.h[0] = *(const v8bf*)(ft + abase + c0 + kb);
                    u.h[1] = *(const v8bf*)(ft + abase + c0 + kb + 8);
                    a = u.v;
                }
#pragma unroll
                for (int nt = 0; nt < 8; ++nt) {
                    const int n = n0 + (nt << 4) + m;
                    v16bf bf = *(const v16bf*)(wt + n * KCONV + c0);
                    acc[nt] = __builtin_amdgcn_wmma_f32_16x16x32_bf16(
                        false, a, false, bf, (short)0, acc[nt], false, false);
                }
            }
        }
    }

    // Epilogue: bias + store bf16 to cf[B][H][W][512]
    const int mrofs = (lane >= 16) ? 8 : 0;
#pragma unroll
    for (int nt = 0; nt < 8; ++nt) {
        const int n = n0 + (nt << 4) + m;
        const float bias = (n < 256) ? coef_b[n] : freq_b[n - 256];
#pragma unroll
        for (int v = 0; v < 8; ++v) {
            const int xs_ = x0 + v + mrofs;
            const int o = (((b << 14) + (y << 7) + xs_) << 9) + n;
            cf[o] = (__bf16)(acc[nt][v] + bias);
        }
    }
}

// ---------------------------------------------------------------------------
// Kernel 4: per-point head. Each wave = 16 rows = 4 points x 4 shifts.
//   Stage 1: gather+trig -> x tile (16x256 bf16 in LDS)
//   Stage 2: 3 WMMA layers (bias+ReLU fused), 128 WMMAs each
//   Stage 3: w3 dot + wave-local area-weighted ensemble + coarse add
// ---------------------------------------------------------------------------
__global__ void __launch_bounds__(128) ph_points(const float* __restrict__ pts,
                                                 const float* __restrict__ cfeat,
                                                 const __bf16* __restrict__ cf,
                                                 const __bf16* __restrict__ Wm,
                                                 const float* __restrict__ b0,
                                                 const float* __restrict__ b1,
                                                 const float* __restrict__ b2,
                                                 const float* __restrict__ w3,
                                                 const float* __restrict__ b3,
                                                 const float* __restrict__ phase_w,
                                                 float* __restrict__ out) {
    __shared__ __attribute__((aligned(32))) __bf16 xs[4][16 * 256];
    __shared__ float s_area[4][16];
    __shared__ float s_pred[4][16][2];

    const int wave = threadIdx.x >> 5;
    const int lane = threadIdx.x & 31;
    const int gw = blockIdx.x * 4 + wave;     // 0..16383
    const int p0 = gw << 2;                   // first point id (over B*Q)
    const float PI = 3.14159265358979f;

    // ---- Stage 1: build 16 activation rows ----
    for (int r = 0; r < 16; ++r) {
        const int pidx = p0 + (r >> 2);
        const int s = r & 3;
        const int b = pidx >> 15;             // Q = 32768
        const int q = pidx & 32767;
        const float pc0 = pts[pidx * 2 + 0];
        const float pc1 = pts[pidx * 2 + 1];
        const float vx = (s >= 2) ? 1.f : -1.f;
        const float vy = (s & 1) ? 1.f : -1.f;
        float c0 = pc0 + vx * (1.f / 128.f) + 1e-6f;
        float c1 = pc1 + vy * (1.f / 128.f) + 1e-6f;
        c0 = fminf(fmaxf(c0, -1.f + 1e-6f), 1.f - 1e-6f);
        c1 = fminf(fmaxf(c1, -1.f + 1e-6f), 1.f - 1e-6f);
        int iy = (int)rintf(((c0 + 1.f) * 128.f - 1.f) * 0.5f);
        int ix = (int)rintf(((c1 + 1.f) * 128.f - 1.f) * 0.5f);
        iy = iy < 0 ? 0 : (iy > 127 ? 127 : iy);
        ix = ix < 0 ? 0 : (ix > 127 ? 127 : ix);
        const float qy = -1.f + 1.f / 128.f + (2.f / 128.f) * (float)iy;
        const float qx = -1.f + 1.f / 128.f + (2.f / 128.f) * (float)ix;
        const float rel0 = (pc0 - qy) * 128.f;
        const float rel1 = (pc1 - qx) * 128.f;
        // faithful "cell" quirk: only q==0 / q==1 rows are scaled (Hc=Wc=32)
        const float cellf = (q <= 1) ? (2.f / 32.f) : 1.f;
        const float rc0 = cellf * 128.f, rc1 = cellf * 128.f;

        const int base = ((b << 14) + (iy << 7) + ix) << 9;   // *512 channels
        const int j0 = lane << 2;                             // 4 j's per lane
        const v8bf fr = *(const v8bf*)(cf + base + 256 + 2 * j0); // freq pairs
        const v4bf cl = *(const v4bf*)(cf + base + j0);           // coef[j]
        const v4bf chv = *(const v4bf*)(cf + base + 128 + j0);    // coef[j+128]
        v4bf xlo, xhi;
#pragma unroll
        for (int jj = 0; jj < 4; ++jj) {
            const int j = j0 + jj;
            const float f0 = (float)fr[2 * jj], f1 = (float)fr[2 * jj + 1];
            const float ph = rc0 * phase_w[2 * j] + rc1 * phase_w[2 * j + 1];
            const float qf = f0 * rel0 + f1 * rel1 + ph;
            const float cc = __cosf(PI * qf);
            const float ss = __sinf(PI * qf);
            xlo[jj] = (__bf16)((float)cl[jj] * cc);
            xhi[jj] = (__bf16)((float)chv[jj] * ss);
        }
        *(v4bf*)(&xs[wave][r * 256 + j0]) = xlo;
        *(v4bf*)(&xs[wave][r * 256 + 128 + j0]) = xhi;
        if (lane == 0) s_area[wave][r] = fabsf(rel0 * rel1) + 1e-9f;
    }
    __syncthreads();

    // ---- Stage 2: three 256x256 WMMA layers with fused bias+ReLU ----
    const int m = lane & 15;
    const int kb  = (lane < 16) ? 0 : 16;
    const int kb2 = (lane < 16) ? 0 : 16;
    const int mrofs = (lane >= 16) ? 8 : 0;
    for (int layer = 0; layer < 3; ++layer) {
        const __bf16* W = Wm + layer * 65536;
        const float* bias = (layer == 0) ? b0 : (layer == 1) ? b1 : b2;
        v16bf afr[8];
#pragma unroll
        for (int kc = 0; kc < 8; ++kc) {
            union { v16bf v; v8bf h[2]; } u;
            u.h[0] = *(const v8bf*)(&xs[wave][m * 256 + kc * 32 + kb]);
            u.h[1] = *(const v8bf*)(&xs[wave][m * 256 + kc * 32 + kb + 8]);
            afr[kc] = u.v;
        }
        for (int nt = 0; nt < 16; ++nt) {
            v8f acc = {};
            const int n = (nt << 4) + m;
#pragma unroll
            for (int kc = 0; kc < 8; ++kc) {
                v16bf bf = *(const v16bf*)(W + n * 256 + kc * 32 + kb2);
                acc = __builtin_amdgcn_wmma_f32_16x16x32_bf16(
                    false, afr[kc], false, bf, (short)0, acc, false, false);
            }
            const float bv = bias[n];
#pragma unroll
            for (int v = 0; v < 8; ++v) {
                float h = acc[v] + bv;
                h = h > 0.f ? h : 0.f;
                xs[wave][(v + mrofs) * 256 + n] = (__bf16)h;
            }
        }
        __syncthreads();
    }

    // ---- Stage 3: tiny final layer (256 -> 2) ----
    {
        const int r = lane & 15, o = lane >> 4;
        float accv = 0.f;
        for (int k = 0; k < 256; k += 8) {
            v8bf hv = *(const v8bf*)(&xs[wave][r * 256 + k]);
#pragma unroll
            for (int kk = 0; kk < 8; ++kk)
                accv += (float)hv[kk] * w3[(k + kk) * 2 + o];
        }
        s_pred[wave][r][o] = accv + b3[o];
    }
    __syncthreads();

    // ---- Combine: diagonal-swapped area weights + coarse features ----
    if (lane < 8) {
        const int pi = lane >> 1, o = lane & 1;
        const int pidx = p0 + pi;
        const int b = pidx >> 15, q = pidx & 32767;
        const float a0 = s_area[wave][pi * 4 + 0];
        const float a1 = s_area[wave][pi * 4 + 1];
        const float a2 = s_area[wave][pi * 4 + 2];
        const float a3 = s_area[wave][pi * 4 + 3];
        const float inv = 1.f / (a0 + a1 + a2 + a3);
        float r = s_pred[wave][pi * 4 + 0][o] * (a3 * inv)
                + s_pred[wave][pi * 4 + 1][o] * (a2 * inv)
                + s_pred[wave][pi * 4 + 2][o] * (a1 * inv)
                + s_pred[wave][pi * 4 + 3][o] * (a0 * inv);
        const int oi = ((b * 2 + o) << 15) + q;
        out[oi] = r + cfeat[oi];
    }
}

// ---------------------------------------------------------------------------
extern "C" void kernel_launch(void* const* d_in, const int* in_sizes, int n_in,
                              void* d_out, int out_size, void* d_ws, size_t ws_size,
                              hipStream_t stream) {
    const float* feat     = (const float*)d_in[0];
    const float* pts      = (const float*)d_in[1];
    const float* cfeat    = (const float*)d_in[2];
    // d_in[3] coarse_sem_seg_logits: only its 32x32 shape is used (hardcoded)
    const float* coef_w   = (const float*)d_in[4];
    const float* coef_b   = (const float*)d_in[5];
    const float* freq_w   = (const float*)d_in[6];
    const float* freq_b   = (const float*)d_in[7];
    const float* phase_w  = (const float*)d_in[8];
    const float* w0 = (const float*)d_in[9];   const float* b0 = (const float*)d_in[10];
    const float* w1 = (const float*)d_in[11];  const float* b1 = (const float*)d_in[12];
    const float* w2 = (const float*)d_in[13];  const float* b2 = (const float*)d_in[14];
    const float* w3 = (const float*)d_in[15];  const float* b3 = (const float*)d_in[16];
    float* out = (float*)d_out;

    // Workspace carve-up (all 32B aligned)
    char* ws = (char*)d_ws;
    __bf16* Wn = (__bf16*)(ws);                              //  2,359,296 B
    __bf16* Wm = (__bf16*)(ws + 2359296);                    //    393,216 B
    __bf16* ft = (__bf16*)(ws + 2359296 + 393216);           // 16,777,216 B
    __bf16* cf = (__bf16*)(ws + 2359296 + 393216 + 16777216);// 33,554,432 B

    // 1) weight packing: 1,376,256 threads
    ph_pack_weights<<<5376, 256, 0, stream>>>(coef_w, freq_w, w0, w1, w2, Wn, Wm);
    // 2) feat transpose/convert: 8,388,608 threads
    ph_feat_t<<<32768, 256, 0, stream>>>(feat, ft);
    // 3) fused conv GEMM: 2048 blocks x 128 threads
    ph_conv<<<2048, 128, 0, stream>>>(ft, Wn, coef_b, freq_b, cf);
    // 4) point head: 4096 blocks x 128 threads (16384 waves x 16 rows)
    ph_points<<<4096, 128, 0, stream>>>(pts, cfeat, cf, Wm, b0, b1, b2,
                                        w3, b3, phase_w, out);
}
